// PensionLensGNN_1168231104685
// MI455X (gfx1250) — matrix operand order repaired and checked by reading, not compiled, not run on hardware
//
#include <hip/hip_runtime.h>
#include <hip/hip_bf16.h>
#include <cstdint>

// ---------------------------------------------------------------------------
// PensionLens hetero-GraphSAGE inference for MI455X (gfx1250, wave32, WMMA)
//
// Precision strategy: all scatter/aggregation math stays f32 (atomics require
// it; means are unit-scale post-ReLU activations).  Dense H=128 GEMMs run on
// the matrix units via v_wmma_f32_16x16x32_f16 with f32 accumulation; weights
// and the 16-row activation tiles are converted to f16 (one-time / in LDS).
// ---------------------------------------------------------------------------

typedef _Float16 h16;
typedef __attribute__((ext_vector_type(16))) _Float16 v16h;
typedef __attribute__((ext_vector_type(8)))  float    v8f;

#define HDIM 128

// ---------------------------- utility kernels ------------------------------

__global__ void fill_zero_kernel(float* __restrict__ p, size_t n) {
  size_t i = (size_t)blockIdx.x * blockDim.x + threadIdx.x;
  size_t stride = (size_t)gridDim.x * blockDim.x;
  for (; i < n; i += stride) p[i] = 0.0f;
}

__global__ void f32_to_f16_kernel(const float* __restrict__ s, h16* __restrict__ d, int n) {
  int i = blockIdx.x * blockDim.x + threadIdx.x;
  if (i < n) d[i] = (h16)s[i];
}

// relu(X[N,K] @ W[H,K]^T + b), K <= 8, one 128-thread block per 16 nodes.
__global__ void input_proj_kernel(const float* __restrict__ X,
                                  const float* __restrict__ W,
                                  const float* __restrict__ b,
                                  float* __restrict__ out, int N, int K) {
  __shared__ float sW[HDIM * 8];
  const int t = threadIdx.x;                 // output channel
  for (int i = t; i < HDIM * K; i += HDIM) sW[i] = W[i];
  const float bias = b[t];
  __syncthreads();
  const int base = blockIdx.x * 16;
  for (int r = 0; r < 16; ++r) {
    const int n = base + r;
    if (n >= N) break;
    float acc = bias;
    #pragma unroll 8
    for (int k = 0; k < K; ++k) acc += X[(size_t)n * K + k] * sW[t * K + k];
    out[(size_t)n * HDIM + t] = fmaxf(acc, 0.0f);
  }
}

__global__ void degree_kernel(const int* __restrict__ src, const int* __restrict__ dst,
                              float* __restrict__ degf, float* __restrict__ degm, int E) {
  int e = blockIdx.x * blockDim.x + threadIdx.x;
  if (e < E) {
    atomicAdd(&degf[src[e]], 1.0f);
    atomicAdd(&degm[dst[e]], 1.0f);
  }
}

__global__ void recip_kernel(float* __restrict__ d, int n) {
  int i = blockIdx.x * blockDim.x + threadIdx.x;
  if (i < n) d[i] = 1.0f / fmaxf(d[i], 1.0f);
}

// One wave per edge; each lane moves float4 and issues 4 f32 L2 atomics.
__global__ void scatter_add_kernel(const float* __restrict__ Xsrc,
                                   const int* __restrict__ idx_src,
                                   const int* __restrict__ idx_dst,
                                   float* __restrict__ agg, int E) {
  size_t gid = (size_t)blockIdx.x * blockDim.x + threadIdx.x;
  size_t e = gid >> 5;
  int lane = (int)(gid & 31);
  if (e >= (size_t)E) return;
  const int s = idx_src[e];
  const int d = idx_dst[e];
  const float4 v = *(const float4*)(Xsrc + (size_t)s * HDIM + lane * 4);
  float* p = agg + (size_t)d * HDIM + lane * 4;
  atomicAdd(p + 0, v.x);
  atomicAdd(p + 1, v.y);
  atomicAdd(p + 2, v.z);
  atomicAdd(p + 3, v.w);
}

// ------------------------- WMMA SAGE combine -------------------------------
// out[n][o] = relu( bias[o] + (Aagg[n]*inv_deg[n]) @ Wl^T  (+ Xroot[n] @ Wr^T) )
// Hin fixed at 128; Hout in {128,64}. Launch: grid=(N+15)/16, block=32*(Hout/16).
// A fragment layout (16x32 f16, contiguous k order used here): lane l: M=l&15, g=l>>4;
//   elem i: K = g*16 + i
// B fragment layout (32x16 f16): lane l: N=l&15, g=l>>4; elem i: K = g*16 + i
// C/D (16x16 f32): VGPR r: M = g*8 + r, N = l&15.
__global__ void sage_combine_wmma(const float* __restrict__ Aagg,
                                  const float* __restrict__ inv_deg,
                                  const float* __restrict__ Xroot,
                                  const h16* __restrict__ Wl,
                                  const float* __restrict__ bias,
                                  const h16* __restrict__ Wr,
                                  float* __restrict__ out,
                                  int N, int Hout) {
  __shared__ h16 sA[16][HDIM + 8];
  __shared__ h16 sR[16][HDIM + 8];
  const int row0 = blockIdx.x * 16;
  const int tid = threadIdx.x;
  const bool has_root = (Xroot != nullptr);

  // Coalesced f32 staging -> f16 LDS tiles (mean scale fused here).
  for (int idx = tid; idx < 16 * HDIM; idx += blockDim.x) {
    const int r = idx >> 7, c = idx & (HDIM - 1);
    const int n = row0 + r;
    float a = 0.0f, x = 0.0f;
    if (n < N) {
      const float s = inv_deg ? inv_deg[n] : 1.0f;
      a = Aagg[(size_t)n * HDIM + c] * s;
      if (has_root) x = Xroot[(size_t)n * HDIM + c];
    }
    sA[r][c] = (h16)a;
    sR[r][c] = (h16)x;
  }
  __syncthreads();

  const int wave = tid >> 5;          // output col-tile owned by this wave
  const int lane = tid & 31;
  const int g = lane >> 4;
  const int m = lane & 15;
  const int ocol = wave * 16 + m;     // weight row / output column

  v8f acc = {};

  #pragma unroll
  for (int kt = 0; kt < HDIM / 32; ++kt) {
    v16h a;
    #pragma unroll
    for (int i = 0; i < 16; ++i) {
      const int v = i >> 1, hh = i & 1;
      const int K = g * 16 + i;
      a[i] = sA[m][kt * 32 + K];
    }
    const v16h b = *(const v16h*)(Wl + (size_t)ocol * HDIM + kt * 32 + g * 16);
    acc = __builtin_amdgcn_wmma_f32_16x16x32_f16(false, a, false, b, (short)0, acc,
                                                 false, false);
  }
  if (has_root) {
    #pragma unroll
    for (int kt = 0; kt < HDIM / 32; ++kt) {
      v16h a;
      #pragma unroll
      for (int i = 0; i < 16; ++i) {
        const int v = i >> 1, hh = i & 1;
        const int K = g * 16 + i;
        a[i] = sR[m][kt * 32 + K];
      }
      const v16h b = *(const v16h*)(Wr + (size_t)ocol * HDIM + kt * 32 + g * 16);
      acc = __builtin_amdgcn_wmma_f32_16x16x32_f16(false, a, false, b, (short)0, acc,
                                                   false, false);
    }
  }

  const float bcol = bias[ocol];
  #pragma unroll
  for (int r = 0; r < 8; ++r) {
    const int n = row0 + g * 8 + r;
    if (n < N) out[(size_t)n * Hout + ocol] = fmaxf(acc[r] + bcol, 0.0f);
  }
}

// logits[n] = h[n,64] @ Wc2[2,64]^T + bc2
__global__ void classifier2_kernel(const float* __restrict__ h,
                                   const float* __restrict__ Wc2,
                                   const float* __restrict__ bc2,
                                   float* __restrict__ out, int N) {
  int n = blockIdx.x * blockDim.x + threadIdx.x;
  if (n >= N) return;
  float a0 = bc2[0], a1 = bc2[1];
  const float* hr = h + (size_t)n * 64;
  #pragma unroll 8
  for (int k = 0; k < 64; ++k) {
    const float x = hr[k];
    a0 += x * Wc2[k];
    a1 += x * Wc2[64 + k];
  }
  out[(size_t)n * 2 + 0] = a0;
  out[(size_t)n * 2 + 1] = a1;
}

// ------------------------------- launcher ----------------------------------

extern "C" void kernel_launch(void* const* d_in, const int* in_sizes, int n_in,
                              void* d_out, int out_size, void* d_ws, size_t ws_size,
                              hipStream_t stream) {
  (void)in_sizes; (void)n_in; (void)out_size;
  constexpr int NF = 100000, NM = 20000, E = 1000000, H = HDIM, FIN = 8, MINF = 5;

  const float* x_fund = (const float*)d_in[0];
  const float* x_mgr  = (const float*)d_in[1];
  const int*   src    = (const int*)d_in[2];   // fm_src (fund ids)
  const int*   dst    = (const int*)d_in[3];   // fm_dst (manager ids)
  const float* Wf = (const float*)d_in[4];  const float* bf = (const float*)d_in[5];
  const float* Wm = (const float*)d_in[6];  const float* bm = (const float*)d_in[7];
  const float* Wl_fm0 = (const float*)d_in[8];
  const float* bl_fm0 = (const float*)d_in[9];
  const float* Wr_fm0 = (const float*)d_in[10];
  const float* Wl_mf0 = (const float*)d_in[11];
  const float* bl_mf0 = (const float*)d_in[12];
  const float* Wr_mf0 = (const float*)d_in[13];
  // d_in[14..16] = Wl_fm1/bl_fm1/Wr_fm1: m2 is deleted in the reference -> dead.
  const float* Wl_mf1 = (const float*)d_in[17];
  const float* bl_mf1 = (const float*)d_in[18];
  const float* Wr_mf1 = (const float*)d_in[19];
  const float* Wc1 = (const float*)d_in[20]; const float* bc1 = (const float*)d_in[21];
  const float* Wc2 = (const float*)d_in[22]; const float* bc2 = (const float*)d_in[23];
  float* out = (float*)d_out;

  // ---- workspace carve (all offsets 256B aligned) ----
  char* base = (char*)d_ws;
  size_t used = 0;
  auto alloc = [&](size_t bytes) -> char* {
    char* p = base + used;
    used += (bytes + 255) & ~(size_t)255;
    return p;
  };
  float* f0  = (float*)alloc((size_t)NF * H * 4);  // f^(0); reused later as f2
  float* m0  = (float*)alloc((size_t)NM * H * 4);
  float* m1  = (float*)alloc((size_t)NM * H * 4);
  float* f1  = (float*)alloc((size_t)NF * H * 4);
  float* agg = (float*)alloc((size_t)NF * H * 4);  // scatter target; reused as classifier hidden
  float* idf = (float*)alloc((size_t)NF * 4);      // inverse degree (fund)
  float* idm = (float*)alloc((size_t)NM * 4);      // inverse degree (manager)
  h16* hWl_fm0 = (h16*)alloc((size_t)H * H * 2);
  h16* hWr_fm0 = (h16*)alloc((size_t)H * H * 2);
  h16* hWl_mf0 = (h16*)alloc((size_t)H * H * 2);
  h16* hWr_mf0 = (h16*)alloc((size_t)H * H * 2);
  h16* hWl_mf1 = (h16*)alloc((size_t)H * H * 2);
  h16* hWr_mf1 = (h16*)alloc((size_t)H * H * 2);
  h16* hWc1    = (h16*)alloc((size_t)64 * H * 2);
  if (used > ws_size) return;  // workspace too small; nothing safe to do

  auto cvt = [&](const float* s, h16* d, int n) {
    f32_to_f16_kernel<<<(n + 255) / 256, 256, 0, stream>>>(s, d, n);
  };
  auto zero = [&](float* p, size_t n) {
    fill_zero_kernel<<<2048, 256, 0, stream>>>(p, n);
  };

  // one-time weight conversion to f16 (WMMA operand format)
  cvt(Wl_fm0, hWl_fm0, H * H);  cvt(Wr_fm0, hWr_fm0, H * H);
  cvt(Wl_mf0, hWl_mf0, H * H);  cvt(Wr_mf0, hWr_mf0, H * H);
  cvt(Wl_mf1, hWl_mf1, H * H);  cvt(Wr_mf1, hWr_mf1, H * H);
  cvt(Wc1, hWc1, 64 * H);

  // input projections
  input_proj_kernel<<<(NF + 15) / 16, HDIM, 0, stream>>>(x_fund, Wf, bf, f0, NF, FIN);
  input_proj_kernel<<<(NM + 15) / 16, HDIM, 0, stream>>>(x_mgr, Wm, bm, m0, NM, MINF);

  // degrees (shared by both layers), then invert in place
  zero(idf, NF); zero(idm, NM);
  degree_kernel<<<(E + 255) / 256, 256, 0, stream>>>(src, dst, idf, idm, E);
  recip_kernel<<<(NF + 255) / 256, 256, 0, stream>>>(idf, NF);
  recip_kernel<<<(NM + 255) / 256, 256, 0, stream>>>(idm, NM);

  const int scatter_blocks = (int)(((size_t)E * 32 + 255) / 256);

  // layer 0, fund -> manager: m1
  zero(agg, (size_t)NM * H);
  scatter_add_kernel<<<scatter_blocks, 256, 0, stream>>>(f0, src, dst, agg, E);
  sage_combine_wmma<<<(NM + 15) / 16, 256, 0, stream>>>(
      agg, idm, m0, hWl_fm0, bl_fm0, hWr_fm0, m1, NM, H);

  // layer 0, manager -> fund: f1
  zero(agg, (size_t)NF * H);
  scatter_add_kernel<<<scatter_blocks, 256, 0, stream>>>(m0, dst, src, agg, E);
  sage_combine_wmma<<<(NF + 15) / 16, 256, 0, stream>>>(
      agg, idf, f0, hWl_mf0, bl_mf0, hWr_mf0, f1, NF, H);

  // layer 1, manager -> fund: f2 (overwrites f0; m2 skipped — dead in reference)
  zero(agg, (size_t)NF * H);
  scatter_add_kernel<<<scatter_blocks, 256, 0, stream>>>(m1, dst, src, agg, E);
  sage_combine_wmma<<<(NF + 15) / 16, 256, 0, stream>>>(
      agg, idf, f1, hWl_mf1, bl_mf1, hWr_mf1, f0, NF, H);

  // classifier hidden: h (reuses agg) = relu(f2 @ Wc1^T + bc1), Hout=64, 4 waves
  sage_combine_wmma<<<(NF + 15) / 16, 128, 0, stream>>>(
      f0, nullptr, nullptr, hWc1, bc1, nullptr, agg, NF, 64);

  // logits
  classifier2_kernel<<<(NF + 255) / 256, 256, 0, stream>>>(agg, Wc2, bc2, out, NF);
}
